// TripletLoss_45037027066265
// MI455X (gfx1250) — matrix operand order repaired and checked by reading, not compiled, not run on hardware
//
#include <hip/hip_runtime.h>

// ---------------------------------------------------------------------------
// Batch-hard triplet loss, fused:  dist = sqrt(clip(sq_i + sq_j - 2*x@x^T))
// - Gram via V_WMMA_F32_16X16X32_F16 with hi/lo f16 split (~fp32 accuracy),
//   three independent accumulator chains (HH/HL/LH) for matrix-pipe ILP.
// - Mining on SQUARED distances (sqrt/clip monotone; order preserved).
// - SYMMETRY: only upper-triangle block tiles are computed; off-diagonal
//   tiles mine BOTH axes (row-side + transposed col-side), halving WMMA work.
// - Branch-free epilogue: pos/neg masking via +/-inf cndmask shared between
//   the two mining axes (neutral elements, so diag blocks need no guards).
// ---------------------------------------------------------------------------

#define D       128
#define MARGIN  0.3f

typedef __attribute__((ext_vector_type(16))) _Float16 v16h;
typedef __attribute__((ext_vector_type(8)))  _Float16 v8h;
typedef __attribute__((ext_vector_type(8)))  float    v8f;

// ---------------------------------------------------------------------------
// Kernel 1: per-row fp32 norms, f16 hi/lo split of embeddings, init ap/an.
// One wave (32 lanes) per row; D=128 -> 4 elements per lane.
// ---------------------------------------------------------------------------
__global__ __launch_bounds__(256)
void prep_kernel(const float* __restrict__ x,
                 _Float16* __restrict__ Ahi, _Float16* __restrict__ Alo,
                 float* __restrict__ sq,
                 unsigned int* __restrict__ apBits,
                 unsigned int* __restrict__ anBits, int n)
{
    const int tid  = threadIdx.x;
    const int lane = tid & 31;
    const int wave = tid >> 5;
    const int row  = blockIdx.x * 8 + wave;

    if (row < n) {
        const float* xr = x + (size_t)row * D;
        float s = 0.0f;
        #pragma unroll
        for (int c = lane; c < D; c += 32) {
            float v = xr[c];
            _Float16 h = (_Float16)v;              // hi part (RTE)
            _Float16 l = (_Float16)(v - (float)h); // residual
            Ahi[(size_t)row * D + c] = h;
            Alo[(size_t)row * D + c] = l;
            s += v * v;
        }
        #pragma unroll
        for (int m = 16; m >= 1; m >>= 1) s += __shfl_xor(s, m, 32);
        if (lane == 0) sq[row] = s;
    }

    // init mining buffers (squared distances are >= 0, so monotone int-bit
    // atomics are valid; an starts at +inf -> rows with no negative give 0).
    int gid = blockIdx.x * 256 + tid;
    if (gid < n) {
        apBits[gid] = 0u;
        anBits[gid] = 0x7F800000u; // +inf
    }
}

// ---------------------------------------------------------------------------
// Kernel 2: 128x128 Gram tile per block (8 waves), K = 128 (whole depth).
// Upper-triangle tiles only (blockIdx.x = rowBlk <= blockIdx.y = colBlk).
// B-tile (hi+lo) staged in LDS; per-wave column-mining partials in a separate
// 8 KB LDS region (72 KB total of the 320 KB WGP pool).
// ---------------------------------------------------------------------------
__global__ __launch_bounds__(256)
void gram_mine_kernel(const _Float16* __restrict__ Ahi,
                      const _Float16* __restrict__ Alo,
                      const float* __restrict__ sq,
                      const int*  __restrict__ tgt,
                      unsigned int* __restrict__ apBits,
                      unsigned int* __restrict__ anBits)
{
    const int rowBlk = blockIdx.x;          // 0..63
    const int colBlk = blockIdx.y;          // 0..63
    if (rowBlk > colBlk) return;            // symmetry: upper triangle only
    const bool diag = (rowBlk == colBlk);

    __shared__ _Float16 Bh[128 * D];        // 32 KB
    __shared__ _Float16 Bl[128 * D];        // 32 KB
    __shared__ float colP[8 * 128];         // 4 KB per-wave col pos-max partials
    __shared__ float colN[8 * 128];         // 4 KB per-wave col neg-min partials

    const int tid    = threadIdx.x;
    const int wave   = tid >> 5;
    const int lane   = tid & 31;
    const int l16    = lane & 15;
    const int half16 = lane >> 4;           // 0: lanes 0-15, 1: lanes 16-31
    const int colBase = colBlk * 128;

    const float NINF = -__builtin_inff();
    const float PINF =  __builtin_inff();

    // ---- stage B tile (rows colBase..+127, all 128 K) : contiguous 32KB x2
    {
        const uint4* gH = (const uint4*)(Ahi + (size_t)colBase * D);
        const uint4* gL = (const uint4*)(Alo + (size_t)colBase * D);
        uint4* sH = (uint4*)Bh;
        uint4* sL = (uint4*)Bl;
        const int total = 128 * (D / 8);    // 2048 x 16B per array
        for (int i = tid; i < total; i += 256) {
            sH[i] = gH[i];
            sL[i] = gL[i];
        }
    }

    // ---- A fragments in registers (16x32 per step, contiguous k order, same as B):
    // lane<16 holds M=l16, K = k0..k0+15 (contiguous);
    // lane>=16 holds M=l16, K = 16..31 of the step.
    const int rowA = rowBlk * 128 + wave * 16 + l16;
    const _Float16* rowH = Ahi + (size_t)rowA * D;
    const _Float16* rowL = Alo + (size_t)rowA * D;
    v16h aH[4], aL[4];
    #pragma unroll
    for (int kc = 0; kc < 4; ++kc) {
        const int k0 = kc * 32 + half16 * 16;
        v8h h0 = *(const v8h*)(rowH + k0);
        v8h h1 = *(const v8h*)(rowH + k0 + 8);
        v8h l0 = *(const v8h*)(rowL + k0);
        v8h l1 = *(const v8h*)(rowL + k0 + 8);
        aH[kc] = __builtin_shufflevector(h0, h1, 0,1,2,3,4,5,6,7,8,9,10,11,12,13,14,15);
        aL[kc] = __builtin_shufflevector(l0, l1, 0,1,2,3,4,5,6,7,8,9,10,11,12,13,14,15);
    }

    // C/D layout: lane -> column (l16), VGPR r -> row r (+8 for upper half).
    const int rowCbase = rowBlk * 128 + wave * 16 + half16 * 8;
    float sqRow[8];
    int   tRow[8];
    #pragma unroll
    for (int r = 0; r < 8; ++r) {
        sqRow[r] = sq[rowCbase + r];
        tRow[r]  = tgt[rowCbase + r];
    }
    // preload this lane's per-subtile column norm/target (one col per subtile)
    float sqCol[8];
    int   tCol[8];
    #pragma unroll
    for (int ct = 0; ct < 8; ++ct) {
        const int gc = colBase + ct * 16 + l16;
        sqCol[ct] = sq[gc];
        tCol[ct]  = tgt[gc];
    }

    float pmax[8], nmin[8];                 // row-side mining state (d^2)
    #pragma unroll
    for (int r = 0; r < 8; ++r) { pmax[r] = 0.0f; nmin[r] = PINF; }

    __syncthreads();

    // ---- 8 column subtiles of 16
    #pragma unroll
    for (int ct = 0; ct < 8; ++ct) {
        // B 32x16, contiguous k order (as A): lane<16 -> N=l16, K=0..15 (contiguous row of
        // X in LDS); lane>=16 -> same N, K=16..31.
        const int colLocal = ct * 16 + l16;
        const _Float16* bHrow = Bh + (size_t)colLocal * D;
        const _Float16* bLrow = Bl + (size_t)colLocal * D;

        v8f acc_hh = {};
        v8f acc_hl = {};
        v8f acc_lh = {};
        #pragma unroll
        for (int kc = 0; kc < 4; ++kc) {
            const int k0 = kc * 32 + half16 * 16;
            v16h bH = *(const v16h*)(bHrow + k0);
            v16h bL = *(const v16h*)(bLrow + k0);
            acc_hh = __builtin_amdgcn_wmma_f32_16x16x32_f16(false, aH[kc], false, bH,
                                                            (short)0, acc_hh, false, false);
            acc_hl = __builtin_amdgcn_wmma_f32_16x16x32_f16(false, aH[kc], false, bL,
                                                            (short)0, acc_hl, false, false);
            acc_lh = __builtin_amdgcn_wmma_f32_16x16x32_f16(false, aL[kc], false, bH,
                                                            (short)0, acc_lh, false, false);
        }

        // ---- fused squared-distance + batch-hard mining, both axes,
        // branch-free: masked values via +/-inf neutral elements, shared
        // between the row-side and col-side updates.
        const float sqc = sqCol[ct];
        const int   tc  = tCol[ct];
        float cp = NINF;                    // col-side pos max (this lane's col)
        float cn = PINF;                    // col-side neg min
        #pragma unroll
        for (int r = 0; r < 8; ++r) {
            float dot = (acc_hh[r] + acc_hl[r]) + acc_lh[r];
            float d2  = sqRow[r] + sqc - 2.0f * dot;
            bool pos  = (tRow[r] == tc);
            float pd2 = pos ? d2 : NINF;    // neutral for max
            float nd2 = pos ? PINF : d2;    // neutral for min
            pmax[r] = fmaxf(pmax[r], pd2);
            nmin[r] = fminf(nmin[r], nd2);
            cp = fmaxf(cp, pd2);
            cn = fminf(cn, nd2);
        }
        // combine the two lane-halves holding the same column; store per-wave
        // partials (diag blocks write too, but never read them back).
        cp = fmaxf(cp, __shfl_xor(cp, 16, 32));
        cn = fminf(cn, __shfl_xor(cn, 16, 32));
        if (half16 == 0) {
            colP[wave * 128 + ct * 16 + l16] = cp;
            colN[wave * 128 + ct * 16 + l16] = cn;
        }
    }

    // ---- row-side: reduce across the 16 lanes sharing the same 8 rows
    #pragma unroll
    for (int s = 1; s < 16; s <<= 1) {
        #pragma unroll
        for (int r = 0; r < 8; ++r) {
            pmax[r] = fmaxf(pmax[r], __shfl_xor(pmax[r], s, 32));
            nmin[r] = fminf(nmin[r], __shfl_xor(nmin[r], s, 32));
        }
    }
    if (l16 == 0) {
        #pragma unroll
        for (int r = 0; r < 8; ++r) {
            atomicMax((int*)&apBits[rowCbase + r], __float_as_int(pmax[r]));
            atomicMin((int*)&anBits[rowCbase + r], __float_as_int(nmin[r]));
        }
    }

    // ---- col-side (transposed tile): combine the 8 wave partials per column
    // (block-uniform guard; skipped entirely on diagonal tiles)
    if (!diag) {
        __syncthreads();
        if (tid < 128) {
            float m = colP[tid];
            #pragma unroll
            for (int w = 1; w < 8; ++w) m = fmaxf(m, colP[w * 128 + tid]);
            // m may be -inf if no positive in this tile; clamp to 0 so the
            // int-bit atomicMax stays on nonnegative floats.
            m = fmaxf(m, 0.0f);
            atomicMax((int*)&apBits[colBase + tid], __float_as_int(m));
        } else {
            const int c = tid - 128;
            float m = colN[c];
            #pragma unroll
            for (int w = 1; w < 8; ++w) m = fminf(m, colN[w * 128 + c]);
            atomicMin((int*)&anBits[colBase + c], __float_as_int(m));
        }
    }
}

// ---------------------------------------------------------------------------
// Kernel 3: loss = mean(relu(sqrt(clip(ap2)) - sqrt(clip(an2)) + margin));
// an2 == +inf rows contribute 0 (matches reference's masked-min semantics).
// ---------------------------------------------------------------------------
__global__ __launch_bounds__(256)
void loss_kernel(const unsigned int* __restrict__ apBits,
                 const unsigned int* __restrict__ anBits,
                 float* __restrict__ out, int n)
{
    __shared__ float red[8];
    const int tid = threadIdx.x;
    float s = 0.0f;
    for (int r = tid; r < n; r += 256) {
        float ap2 = __int_as_float(apBits[r]);
        float an2 = __int_as_float(anBits[r]);
        float ap  = sqrtf(fmaxf(ap2, 1e-12f));
        float an  = sqrtf(fmaxf(an2, 1e-12f));
        float v   = ap - an + MARGIN;
        s += (v > 0.0f) ? v : 0.0f;
    }
    #pragma unroll
    for (int m = 16; m >= 1; m >>= 1) s += __shfl_xor(s, m, 32);
    if ((tid & 31) == 0) red[tid >> 5] = s;
    __syncthreads();
    if (tid == 0) {
        float t = 0.0f;
        #pragma unroll
        for (int w = 0; w < 8; ++w) t += red[w];
        out[0] = t / (float)n;
    }
}

// ---------------------------------------------------------------------------
extern "C" void kernel_launch(void* const* d_in, const int* in_sizes, int n_in,
                              void* d_out, int out_size, void* d_ws, size_t ws_size,
                              hipStream_t stream)
{
    const float* x   = (const float*)d_in[0];   // [N, 128] fp32
    const int*   tgt = (const int*)d_in[1];     // [N] int
    float* out = (float*)d_out;                 // scalar loss
    const int n = in_sizes[1];                  // N = 8192

    // workspace layout (16B-aligned slices)
    char* ws = (char*)d_ws;
    float*        sqv = (float*)ws;                              // n floats
    unsigned int* ap  = (unsigned int*)(ws + (size_t)n * 4);     // n u32
    unsigned int* an  = (unsigned int*)(ws + (size_t)n * 8);     // n u32
    _Float16*     Ahi = (_Float16*)(ws + (size_t)n * 12);        // n*D f16
    _Float16*     Alo = Ahi + (size_t)n * D;                     // n*D f16

    prep_kernel<<<(n + 7) / 8, 256, 0, stream>>>(x, Ahi, Alo, sqv, ap, an, n);

    dim3 grid(n / 128, n / 128);                // lower-triangle blocks early-exit
    gram_mine_kernel<<<grid, 256, 0, stream>>>(Ahi, Alo, sqv, tgt, ap, an);

    loss_kernel<<<1, 256, 0, stream>>>(ap, an, out, n);
}
